// PatchNCELoss_53163105190032
// MI455X (gfx1250) — matrix operand order repaired and checked by reading, not compiled, not run on hardware
//
#include <hip/hip_runtime.h>
#include <hip/hip_bf16.h>

typedef __attribute__((ext_vector_type(16))) __bf16        v16bf;
typedef __attribute__((ext_vector_type(8)))  float         v8f;
typedef __attribute__((ext_vector_type(4)))  unsigned int  v4u;

union F16x16 {
    v16bf          bf;
    v4u            u4[2];
    unsigned short s[16];
};

// Problem constants (from setup_inputs)
constexpr int B_  = 16;
constexpr int C_  = 256;
constexpr int NQ  = 2048;
constexpr int NN  = 2048;
constexpr int RS  = 264;          // padded LDS row stride (halves): 528B -> 4-bank skew per row
constexpr float INV_T  = 10.0f;
constexpr float LOG2E  = 1.4426950408889634f;
constexpr float LN2    = 0.6931471805599453f;

// Packed-fragment geometry: nhi/nlo[b][tile(128)][chunk(8)][lane(32)][e(16)] bf16
constexpr int    FRAG_ELEMS   = 512;                       // 32 lanes * 16 elems per chunk
constexpr size_t N_ELEMS      = (size_t)B_ * C_ * NN;      // 8,388,608
constexpr size_t WS_NEEDED    = 2 * N_ELEMS * sizeof(unsigned short); // 32 MB

// Split f32 into bf16 hi (truncate) + bf16 lo (truncated residual); hi+lo ~ 16 mantissa bits.
__device__ __forceinline__ void bf16split(float x, unsigned short& hs, unsigned short& ls) {
    unsigned u = __float_as_uint(x);
    hs = (unsigned short)(u >> 16);
    float hf = __uint_as_float(u & 0xffff0000u);
    float r  = x - hf;
    ls = (unsigned short)(__float_as_uint(r) >> 16);
}

__global__ void patchnce_zero(float* out) { out[0] = 0.0f; }

// ---------------------------------------------------------------------------
// Pack pass: N f32 -> pre-swizzled bf16 hi/lo B-fragments in d_ws.
// One block per (b, n-tile): stage 256x16 f32 tile in LDS (coalesced reads),
// then each thread emits one fragment lane-slot with two 16B stores (coalesced).
// ---------------------------------------------------------------------------
__global__ __launch_bounds__(256)
void patchnce_pack(const float* __restrict__ nf,
                   unsigned short* __restrict__ nhi,
                   unsigned short* __restrict__ nlo) {
    __shared__ float sN[C_ * 16];                  // [c][ln]
    const int t   = threadIdx.x;
    const int blk = blockIdx.x;                    // b*128 + tile
    const int n0  = (blk & 127) * 16;
    const float* src = nf + (size_t)(blk >> 7) * C_ * NN + n0;
    {
        const int ln = t & 15, cg = t >> 4;
        #pragma unroll 4
        for (int i = 0; i < 16; ++i) {
            const int c = i * 16 + cg;
            sN[c * 16 + ln] = src[(size_t)c * NN + ln];
        }
    }
    __syncthreads();
    const int chunk = t >> 5, lane = t & 31;
    const int h = lane >> 4, ln = lane & 15;
    union { unsigned short s[16]; v4u u[2]; } H, L;
    #pragma unroll
    for (int e = 0; e < 16; ++e) {
        const int c = chunk * 32 + h * 8 + (e & 7) + (e >> 3) * 16;     // K = chunk*32 + h*8 + (e&7) + (e>>3)*16
        bf16split(sN[c * 16 + ln], H.s[e], L.s[e]);
    }
    const size_t dst = ((size_t)blk * 8 + chunk) * FRAG_ELEMS + lane * 16;
    *(v4u*)(nhi + dst)     = H.u[0];
    *(v4u*)(nhi + dst + 8) = H.u[1];
    *(v4u*)(nlo + dst)     = L.u[0];
    *(v4u*)(nlo + dst + 8) = L.u[1];
}

// ---------------------------------------------------------------------------
// Shared prologue/epilogue helpers for the main kernels
// ---------------------------------------------------------------------------
__device__ __forceinline__ void stage_q_and_pos(const float* qf, const float* pf,
                                                int b, int q0, int t,
                                                unsigned short* sQhi, unsigned short* sQlo,
                                                float (*sP)[8]) {
    {   // Q tile (32 rows x 256 C) -> LDS bf16 hi/lo, [m][c] padded layout
        const float* qb = qf + (size_t)b * C_ * NQ + q0;
        const int m = t & 31, cg = t >> 5;
        #pragma unroll 4
        for (int i = 0; i < 32; ++i) {
            const int c = i * 8 + cg;
            unsigned short hs, ls;
            bf16split(qb[(size_t)c * NQ + m], hs, ls);
            sQhi[m * RS + c] = hs;
            sQlo[m * RS + c] = ls;
        }
    }
    {   // positive logits, exact f32: 8 partials per row
        const int row = t >> 3, part = t & 7;
        const float* qb = qf + (size_t)b * C_ * NQ + q0 + row;
        const float* pb = pf + (size_t)b * C_;
        float acc = 0.0f;
        #pragma unroll 8
        for (int cc = 0; cc < 32; ++cc) {
            const int c = part * 32 + cc;
            acc = fmaf(qb[(size_t)c * NQ], pb[c], acc);
        }
        sP[row][part] = acc;
    }
}

__device__ __forceinline__ void epilogue(float mrun[2][8], float lrun[2][8],
                                         int t, int w, int h, int ln,
                                         float (*sRedM)[2][2][8], float (*sRedL)[2][2][8],
                                         float (*sP)[8], float* out) {
    // merge 16 columns of each row-half across lanes (xor tree within half)
    #pragma unroll
    for (int mask = 1; mask <= 8; mask <<= 1) {
        #pragma unroll
        for (int mt = 0; mt < 2; ++mt)
            #pragma unroll
            for (int r = 0; r < 8; ++r) {
                float mo = mrun[mt][r], lo = lrun[mt][r];
                float m2 = __shfl_xor(mo, mask, 32);
                float l2 = __shfl_xor(lo, mask, 32);
                float mn = fmaxf(mo, m2);
                lrun[mt][r] = lo * exp2f(mo - mn) + l2 * exp2f(m2 - mn);
                mrun[mt][r] = mn;
            }
    }
    if (ln == 0) {
        #pragma unroll
        for (int mt = 0; mt < 2; ++mt)
            #pragma unroll
            for (int r = 0; r < 8; ++r) {
                sRedM[w][h][mt][r] = mrun[mt][r];
                sRedL[w][h][mt][r] = lrun[mt][r];
            }
    }
    __syncthreads();
    if (t < 32) {
        const int mt = t >> 4, hh = (t >> 3) & 1, r = t & 7;
        float p = 0.0f;
        #pragma unroll
        for (int j2 = 0; j2 < 8; ++j2) p += sP[t][j2];
        float m = p * (INV_T * LOG2E);             // positive as (m, l=1)
        float l = 1.0f;
        #pragma unroll
        for (int w2 = 0; w2 < 8; ++w2) {
            float mw = sRedM[w2][hh][mt][r];
            float lw = sRedL[w2][hh][mt][r];
            float mn = fmaxf(m, mw);
            l = l * exp2f(m - mn) + lw * exp2f(mw - mn);
            m = mn;
        }
        float loss = LN2 * (m + log2f(l)) - p * INV_T;
        #pragma unroll
        for (int off = 16; off >= 1; off >>= 1)
            loss += __shfl_xor(loss, off, 32);
        if (t == 0) atomicAdd(out, loss * (1.0f / (float)(B_ * NQ)));
    }
}

#define WMMA_BF16(A, Bf, Cacc) \
    __builtin_amdgcn_wmma_f32_16x16x32_bf16(false, (A), false, (Bf), (short)0, (Cacc), false, false)

// ---------------------------------------------------------------------------
// Main kernel, packed-B path: B fragments pre-converted in d_ws.
// Per chunk: 4x global b128 + 8x ds b128 + 6x v_wmma. No conversion VALU.
// ---------------------------------------------------------------------------
__global__ __launch_bounds__(256)
void patchnce_main_packed(const float* __restrict__ qf, const float* __restrict__ pf,
                          const unsigned short* __restrict__ nhi,
                          const unsigned short* __restrict__ nlo,
                          float* __restrict__ out) {
    __shared__ unsigned short sQhi[32 * RS];
    __shared__ unsigned short sQlo[32 * RS];
    __shared__ float sRedM[8][2][2][8];
    __shared__ float sRedL[8][2][2][8];
    __shared__ float sP[32][8];

    const int t  = threadIdx.x;
    const int b  = blockIdx.x >> 6;
    const int q0 = (blockIdx.x & 63) * 32;

    stage_q_and_pos(qf, pf, b, q0, t, sQhi, sQlo, sP);
    __syncthreads();

    const int w = t >> 5, lane = t & 31;
    const int h = lane >> 4, ln = lane & 15;

    float mrun[2][8], lrun[2][8];
    #pragma unroll
    for (int mt = 0; mt < 2; ++mt)
        #pragma unroll
        for (int r = 0; r < 8; ++r) { mrun[mt][r] = -__builtin_inff(); lrun[mt][r] = 0.0f; }

    const float sc = INV_T * LOG2E;

    for (int j = 0; j < 16; ++j) {
        const int tIdx = j * 8 + w;
        const unsigned short* bh = nhi + ((size_t)(b * 128 + tIdx) * 8) * FRAG_ELEMS + lane * 16;
        const unsigned short* bl = nlo + ((size_t)(b * 128 + tIdx) * 8) * FRAG_ELEMS + lane * 16;
        if (j + 1 < 16) {
            __builtin_prefetch(bh + 8 * 8 * FRAG_ELEMS, 0, 0);   // next tile (+8 tiles)
            __builtin_prefetch(bl + 8 * 8 * FRAG_ELEMS, 0, 0);
        }

        v8f acc0 = {}; v8f acc1 = {};
        #pragma unroll
        for (int k = 0; k < 8; ++k) {
            F16x16 bhi, blo;
            bhi.u4[0] = *(const v4u*)(bh + k * FRAG_ELEMS);
            bhi.u4[1] = *(const v4u*)(bh + k * FRAG_ELEMS + 8);
            blo.u4[0] = *(const v4u*)(bl + k * FRAG_ELEMS);
            blo.u4[1] = *(const v4u*)(bl + k * FRAG_ELEMS + 8);
            const int c0 = k * 32 + h * 8;
            #pragma unroll
            for (int mt = 0; mt < 2; ++mt) {
                const unsigned short* ah = &sQhi[(mt * 16 + ln) * RS + c0];
                const unsigned short* al = &sQlo[(mt * 16 + ln) * RS + c0];
                F16x16 Ahi, Alo;
                Ahi.u4[0] = *(const v4u*)(ah); Ahi.u4[1] = *(const v4u*)(ah + 16);
                Alo.u4[0] = *(const v4u*)(al); Alo.u4[1] = *(const v4u*)(al + 16);
                v8f& acc = mt ? acc1 : acc0;
                acc = WMMA_BF16(Ahi.bf, bhi.bf, acc);
                acc = WMMA_BF16(Ahi.bf, blo.bf, acc);
                acc = WMMA_BF16(Alo.bf, bhi.bf, acc);
            }
        }
        #pragma unroll
        for (int mt = 0; mt < 2; ++mt) {
            v8f& acc = mt ? acc1 : acc0;
            #pragma unroll
            for (int r = 0; r < 8; ++r) {
                float s2 = acc[r] * sc;
                float mo = mrun[mt][r];
                float mn = fmaxf(mo, s2);
                lrun[mt][r] = lrun[mt][r] * exp2f(mo - mn) + exp2f(s2 - mn);
                mrun[mt][r] = mn;
            }
        }
    }
    epilogue(mrun, lrun, t, w, h, ln, sRedM, sRedL, sP, out);
}

// ---------------------------------------------------------------------------
// Fallback path (ws too small): fused in-loop conversion (round-1 kernel).
// ---------------------------------------------------------------------------
__global__ __launch_bounds__(256)
void patchnce_main_fused(const float* __restrict__ qf, const float* __restrict__ pf,
                         const float* __restrict__ nf, float* __restrict__ out) {
    __shared__ unsigned short sQhi[32 * RS];
    __shared__ unsigned short sQlo[32 * RS];
    __shared__ float sRedM[8][2][2][8];
    __shared__ float sRedL[8][2][2][8];
    __shared__ float sP[32][8];

    const int t  = threadIdx.x;
    const int b  = blockIdx.x >> 6;
    const int q0 = (blockIdx.x & 63) * 32;

    stage_q_and_pos(qf, pf, b, q0, t, sQhi, sQlo, sP);
    __syncthreads();

    const int w = t >> 5, lane = t & 31;
    const int h = lane >> 4, ln = lane & 15;
    const float* nb = nf + ((size_t)b * C_ + h * 8) * NN + ln;

    float mrun[2][8], lrun[2][8];
    #pragma unroll
    for (int mt = 0; mt < 2; ++mt)
        #pragma unroll
        for (int r = 0; r < 8; ++r) { mrun[mt][r] = -__builtin_inff(); lrun[mt][r] = 0.0f; }

    const float sc = INV_T * LOG2E;

    for (int j = 0; j < 16; ++j) {
        const float* np = nb + (j * 8 + w) * 16;
        v8f acc0 = {}; v8f acc1 = {};
        #pragma unroll 2
        for (int kc = 0; kc < C_; kc += 32) {
            F16x16 bhi, blo;
            #pragma unroll
            for (int e = 0; e < 16; ++e)
                bf16split(np[(size_t)(kc + (e & 7) + (e >> 3) * 16) * NN], bhi.s[e], blo.s[e]);
            const int c0 = kc + h * 8;
            #pragma unroll
            for (int mt = 0; mt < 2; ++mt) {
                const unsigned short* ah = &sQhi[(mt * 16 + ln) * RS + c0];
                const unsigned short* al = &sQlo[(mt * 16 + ln) * RS + c0];
                F16x16 Ahi, Alo;
                Ahi.u4[0] = *(const v4u*)(ah); Ahi.u4[1] = *(const v4u*)(ah + 16);
                Alo.u4[0] = *(const v4u*)(al); Alo.u4[1] = *(const v4u*)(al + 16);
                v8f& acc = mt ? acc1 : acc0;
                acc = WMMA_BF16(Ahi.bf, bhi.bf, acc);
                acc = WMMA_BF16(Ahi.bf, blo.bf, acc);
                acc = WMMA_BF16(Alo.bf, bhi.bf, acc);
            }
        }
        #pragma unroll
        for (int mt = 0; mt < 2; ++mt) {
            v8f& acc = mt ? acc1 : acc0;
            #pragma unroll
            for (int r = 0; r < 8; ++r) {
                float s2 = acc[r] * sc;
                float mo = mrun[mt][r];
                float mn = fmaxf(mo, s2);
                lrun[mt][r] = lrun[mt][r] * exp2f(mo - mn) + exp2f(s2 - mn);
                mrun[mt][r] = mn;
            }
        }
    }
    epilogue(mrun, lrun, t, w, h, ln, sRedM, sRedL, sP, out);
}

extern "C" void kernel_launch(void* const* d_in, const int* in_sizes, int n_in,
                              void* d_out, int out_size, void* d_ws, size_t ws_size,
                              hipStream_t stream) {
    (void)in_sizes; (void)n_in; (void)out_size;
    const float* q  = (const float*)d_in[0];
    const float* p  = (const float*)d_in[1];
    const float* ng = (const float*)d_in[2];
    float* out = (float*)d_out;

    patchnce_zero<<<1, 1, 0, stream>>>(out);

    if (d_ws && ws_size >= WS_NEEDED) {
        unsigned short* nhi = (unsigned short*)d_ws;
        unsigned short* nlo = nhi + N_ELEMS;
        patchnce_pack<<<B_ * (NN / 16), 256, 0, stream>>>(ng, nhi, nlo);
        patchnce_main_packed<<<B_ * (NQ / 32), 256, 0, stream>>>(q, p, nhi, nlo, out);
    } else {
        patchnce_main_fused<<<B_ * (NQ / 32), 256, 0, stream>>>(q, p, ng, out);
    }
}
